// BiSS2D_67611375174267
// MI455X (gfx1250) — 16-bit and 8-bit matrix operand order repaired and checked by reading, compiled, not run on hardware
//
#include <hip/hip_runtime.h>
#include <hip/hip_bf16.h>

typedef _Float16 h16;
typedef __attribute__((ext_vector_type(16))) _Float16 v16h;
typedef __attribute__((ext_vector_type(8)))  float    v8f;
typedef __attribute__((ext_vector_type(8)))  int      v8i;

union AF16 { v16h v; _Float16 h[16]; };
union AF32 { v8f  v; float    f[8]; };
union AI32 { v8i  v; int      i[8]; };

__device__ __forceinline__ float sgnf(float x){ return (x>0.f)?1.f:((x<0.f)?-1.f:0.f); }
__device__ __forceinline__ _Float16 sgnh(float x){ return (_Float16)sgnf(x); }
__device__ __forceinline__ int   sgni(float x){ return (x>0.f) - (x<0.f); }
__device__ __forceinline__ float siluf(float x){ return x / (1.f + __expf(-x)); }
__device__ __forceinline__ float softplusf(float x){ return (x > 20.f) ? x : log1pf(__expf(x)); }

__device__ __forceinline__ int akmap(int i, int hh){
  int v = i >> 1, p = i & 1;
  int base = (v < 4) ? (2*v + p) : (16 + 2*(v-4) + p);
  return base + 8*hh;
}
__device__ __forceinline__ int a8_kbase(int v, int hh){
  int vv = v & 3, hi = (v >> 2) * 32;
  return ((vv < 2) ? (4*vv) : (16 + 4*(vv-2))) + 8*hh + hi;
}
__device__ __forceinline__ int b8_kbase(int v, int hh){
  return a8_kbase(v, hh);
}
__device__ __forceinline__ int map_l(int k, int l){
  if (k & 2) l = 4095 - l;
  if (k & 1) l = ((l & 63) << 6) | (l >> 6);
  return l;
}

#define WMMA_F16(acc,a,b) \
  (acc) = __builtin_amdgcn_wmma_f32_16x16x32_f16(false,(a),false,(b),(short)0,(acc),false,false)
#define WMMA_IU8(acc,a,b) \
  (acc) = __builtin_amdgcn_wmma_i32_16x16x64_iu8(true,(a),true,(b),(acc),false,false)

__global__ void kprep_rowsign_f16(const float* __restrict__ W, h16* __restrict__ out,
                                  int rows, int cols){
  int r = blockIdx.x*blockDim.x + threadIdx.x;
  if (r >= rows) return;
  const float* w = W + (size_t)r*cols;
  float s = 0.f;
  for (int c = 0; c < cols; ++c) s += w[c];
  float mean = s / (float)cols;
  h16* o = out + (size_t)r*cols;
  for (int c = 0; c < cols; ++c) o[c] = sgnh(w[c] - mean);
}

__global__ void kprep_rowsign_i8(const float* __restrict__ W, signed char* __restrict__ out,
                                 int rows, int cols){
  int r = blockIdx.x*blockDim.x + threadIdx.x;
  if (r >= rows) return;
  const float* w = W + (size_t)r*cols;
  float s = 0.f;
  for (int c = 0; c < cols; ++c) s += w[c];
  float mean = s / (float)cols;
  signed char* o = out + (size_t)r*cols;
  for (int c = 0; c < cols; ++c) o[c] = (signed char)sgni(w[c] - mean);
}

__global__ void kprep_conv(const float* __restrict__ cW, signed char* __restrict__ sCW8,
                           float* __restrict__ sc){
  int o = blockIdx.x*blockDim.x + threadIdx.x;
  if (o >= 192) return;
  const float* w = cW + (size_t)o*1728;
  float s = 0.f;
  for (int i = 0; i < 1728; ++i) s += fabsf(w[i]);
  sc[o] = s * (1.f/1728.f);
  signed char* d = sCW8 + (size_t)o*1728;
  for (int i = 0; i < 192; ++i)
    for (int tap = 0; tap < 9; ++tap)
      d[tap*192 + i] = (signed char)sgni(w[i*9 + tap]);
}

__global__ void kprep_wd(const float* __restrict__ Wd, const float* __restrict__ sd,
                         float* __restrict__ wdeff){
  int r = blockIdx.x*blockDim.x + threadIdx.x;
  if (r >= 768) return;
  const float* w = Wd + (size_t)r*6;
  float s = 0.f;
  for (int i = 0; i < 6; ++i) s += w[i];
  float mean = s * (1.f/6.f), sv = sd[r];
  for (int i = 0; i < 6; ++i) wdeff[r*6+i] = sv * sgnf(w[i] - mean);
}

__global__ void k1_in_gemm(const float* __restrict__ x, const h16* __restrict__ sW,
                           const float* __restrict__ s_in, const float* __restrict__ b_in,
                           float* __restrict__ xp, float* __restrict__ z){
  int tile = blockIdx.x;
  int tn = tile % 24, tm = tile / 24;
  int lane = threadIdx.x & 31, hh = lane >> 4, r = lane & 15;
  AF32 acc; v8f zz = {}; acc.v = zz;
  int m = tm*16 + r, n = tn*16 + r;
  #pragma unroll 1
  for (int kc = 0; kc < 3; ++kc){
    AF16 a, bm;
    #pragma unroll
    for (int i = 0; i < 16; ++i){
      int kk = kc*32 + akmap(i, hh);
      a.h[i] = sgnh(x[(size_t)m*96 + kk]);
    }
    #pragma unroll
    for (int i = 0; i < 16; ++i){
      int kk = kc*32 + akmap(i, hh);
      bm.h[i] = sW[(size_t)n*96 + kk];
    }
    WMMA_F16(acc.v, a.v, bm.v);
  }
  float scv = s_in[n], bb = b_in[n];
  #pragma unroll
  for (int j = 0; j < 8; ++j){
    int mm = tm*16 + j + 8*hh;
    int b = mm >> 12, l = mm & 4095;
    float val = acc.f[j]*scv + bb;
    if (n < 192) xp[((size_t)(b*192 + n))*4096 + l] = val;
    else         z [((size_t)(b*4096 + l))*192 + (n - 192)] = val;
  }
}

#define CONV_LDS_I32 (3*18*192/4)
__global__ void k2_conv(const float* __restrict__ xp, const signed char* __restrict__ sCW8,
                        const float* __restrict__ sc, const float* __restrict__ move0,
                        const float* __restrict__ conv_b, const float* __restrict__ rp0,
                        const float* __restrict__ pa, const float* __restrict__ rp1,
                        float* __restrict__ xconv){
  __shared__ int tile_i[CONV_LDS_I32];
  signed char* tile = (signed char*)tile_i;
  int tp = blockIdx.x;
  int b = tp >> 8, rem = tp & 255;
  int y = rem >> 2, x0 = (rem & 3) * 16;
  int tid = threadIdx.x;
  int wave = tid >> 5, lane = tid & 31, hh = lane >> 4, r = lane & 15;
  int o = wave*16 + r;
  __builtin_prefetch(sCW8 + (size_t)o*1728, 0, 0);
  for (int e = tid; e < 3*18*192; e += 384){
    int c = e % 192;
    int t2 = e / 192;
    int xo = t2 % 18, row = t2 / 18;
    int yy = y - 1 + row, xx = x0 - 1 + xo;
    float v = 0.f;
    if (yy >= 0 && yy < 64 && xx >= 0 && xx < 64)
      v = xp[((size_t)(b*192 + c))*4096 + yy*64 + xx] + move0[c];
    tile[e] = (signed char)sgni(v);
  }
  __syncthreads();
  const int* ldsw = (const int*)tile;
  AI32 acc; v8i zzi = {}; acc.v = zzi;
  #pragma unroll 1
  for (int tap = 0; tap < 9; ++tap){
    int dy = tap/3 - 1, dx = tap%3 - 1;
    int rowbase = ((dy+1)*18 + (r + dx + 1)) * 192;
    const int* bw = (const int*)(sCW8 + ((size_t)o*9 + tap)*192);
    #pragma unroll 1
    for (int kc = 0; kc < 3; ++kc){
      AI32 a, bm;
      #pragma unroll
      for (int v = 0; v < 8; ++v)
        a.i[v] = ldsw[(rowbase + kc*64 + a8_kbase(v, hh)) >> 2];
      #pragma unroll
      for (int v = 0; v < 8; ++v)
        bm.i[v] = bw[(kc*64 + b8_kbase(v, hh)) >> 2];
      WMMA_IU8(acc.v, a.v, bm.v);
    }
  }
  float scv = sc[o], cb = conv_b[o], r0v = rp0[o], av = pa[o], r1v = rp1[o];
  #pragma unroll
  for (int j = 0; j < 8; ++j){
    int xx = x0 + j + 8*hh;
    size_t idx = ((size_t)(b*192 + o))*4096 + y*64 + xx;
    float res = xp[idx];
    float v = (float)acc.i[j]*scv + cb + r0v;
    v = (v >= 0.f) ? v : av*v;
    v = v + r1v + res;
    xconv[idx] = siluf(v);
  }
}

__global__ void k3_xdbl(const float* __restrict__ xconv, const signed char* __restrict__ sWl8,
                        const float* __restrict__ sl, const float* __restrict__ bl,
                        float* __restrict__ xdbl){
  int tile = blockIdx.x;
  int tn = tile % 3, tm = tile / 3;
  int mg0 = tm * 16;
  int b = mg0 >> 14, k = (mg0 >> 12) & 3;
  int lane = threadIdx.x & 31, hh = lane >> 4, r = lane & 15;
  int m = (mg0 & 4095) + r;
  AI32 acc; v8i zzi = {}; acc.v = zzi;
  const float* xcb = xconv + (size_t)b*192*4096;
  int n = tn*16 + r;
  const int* bw = (const int*)(sWl8 + (size_t)(k*38 + (n < 38 ? n : 0))*192);
  #pragma unroll 1
  for (int kc = 0; kc < 3; ++kc){
    AI32 a, bm;
    #pragma unroll
    for (int v = 0; v < 8; ++v){
      int word = 0;
      #pragma unroll
      for (int p = 0; p < 4; ++p){
        int kk = kc*64 + a8_kbase(v, hh) + p;
        int f = m*192 + kk;
        int d = f >> 12, l0 = f & 4095;
        int s = sgni(xcb[(size_t)d*4096 + map_l(k, l0)]);
        word |= (s & 0xff) << (8*p);
      }
      a.i[v] = word;
    }
    #pragma unroll
    for (int v = 0; v < 8; ++v)
      bm.i[v] = (n < 38) ? bw[(kc*64 + b8_kbase(v, hh)) >> 2] : 0;
    WMMA_IU8(acc.v, a.v, bm.v);
  }
  if (n < 38){
    float scv = sl[k*38 + n], bb = bl[k*38 + n];
    #pragma unroll
    for (int j = 0; j < 8; ++j){
      int mm = (mg0 & 4095) + j + 8*hh;
      xdbl[((size_t)((b*4 + k)*38 + n))*4096 + mm] = (float)acc.i[j]*scv + bb;
    }
  }
}

__global__ void k4_dts(const float* __restrict__ xdbl, const float* __restrict__ wdeff,
                       const float* __restrict__ bd, const float* __restrict__ dt_bias,
                       float* __restrict__ dts){
  int idx = blockIdx.x*blockDim.x + threadIdx.x;
  int l = idx & 4095;
  int rest = idx >> 12;
  int o = rest % 192;
  int bk = rest / 192;
  int k = bk & 3;
  const float* xb = xdbl + (size_t)bk*38*4096;
  float s = 0.f;
  #pragma unroll
  for (int rr = 0; rr < 6; ++rr){
    int f = l*6 + rr;
    s += sgnf(xb[(size_t)(f >> 12)*4096 + (f & 4095)]) * wdeff[(k*192 + o)*6 + rr];
  }
  s += bd[k*192 + o];
  dts[idx] = softplusf(s + dt_bias[k*192 + o]);
}

__global__ void k5_zero(float* __restrict__ y){
  int idx = blockIdx.x*blockDim.x + threadIdx.x;
  y[idx] = 0.f;
}

__global__ void k6_scan(const float* __restrict__ dts, const float* __restrict__ xdbl,
                        const float* __restrict__ xconv, const float* __restrict__ A_logs,
                        const float* __restrict__ Ds, float* __restrict__ ycomb){
  int chain = blockIdx.x*16 + (threadIdx.x >> 4);
  int n = threadIdx.x & 15;
  int d = chain % 192;
  int bk = chain / 192;
  int k = bk & 3, b = bk >> 2;
  float a  = -__expf(A_logs[(size_t)(k*192 + d)*16 + n]);
  float Dv = Ds[k*192 + d];
  const float* dtp = dts   + (size_t)chain*4096;
  const float* Bp  = xdbl  + ((size_t)bk*38 + 6  + n)*4096;
  const float* Cp  = xdbl  + ((size_t)bk*38 + 22 + n)*4096;
  const float* xc  = xconv + ((size_t)(b*192 + d))*4096;
  float*       yo  = ycomb + ((size_t)(b*192 + d))*4096;
  float h = 0.f;
  for (int l = 0; l < 4096; ++l){
    int lm = map_l(k, l);
    float u  = xc[lm];
    float dt = dtp[l];
    h = __expf(dt*a)*h + dt*u*Bp[l];
    float y = h * Cp[l];
    y += __shfl_xor(y, 8, 16);
    y += __shfl_xor(y, 4, 16);
    y += __shfl_xor(y, 2, 16);
    y += __shfl_xor(y, 1, 16);
    if (n == 0) atomicAdd(&yo[lm], y + Dv*u);
  }
}

__global__ void k7_ln(const float* __restrict__ ycomb, const float* __restrict__ z,
                      const float* __restrict__ lnw, const float* __restrict__ lnb,
                      float* __restrict__ y2){
  __shared__ float s1[192], s2[192];
  int pix = blockIdx.x;
  int b = pix >> 12, l = pix & 4095;
  int c = threadIdx.x;
  float v = ycomb[((size_t)(b*192 + c))*4096 + l];
  s1[c] = v; s2[c] = v*v;
  __syncthreads();
  for (int s = 96; s > 1; s >>= 1){
    if (c < s){ s1[c] += s1[c + s]; s2[c] += s2[c + s]; }
    __syncthreads();
  }
  float mean = (s1[0] + s1[1] + s1[2]) * (1.f/192.f);
  float ex2  = (s2[0] + s2[1] + s2[2]) * (1.f/192.f);
  float inv  = rsqrtf(ex2 - mean*mean + 1e-5f);
  float zn   = z[(size_t)pix*192 + c];
  y2[(size_t)pix*192 + c] = ((v - mean)*inv*lnw[c] + lnb[c]) * siluf(zn);
}

__global__ void k8_out(const float* __restrict__ y2, const signed char* __restrict__ sWo8,
                       const float* __restrict__ s_out, const float* __restrict__ b_out,
                       float* __restrict__ out){
  int tile = blockIdx.x;
  int tn = tile % 6, tm = tile / 6;
  int lane = threadIdx.x & 31, hh = lane >> 4, r = lane & 15;
  AI32 acc; v8i zzi = {}; acc.v = zzi;
  int m = tm*16 + r, n = tn*16 + r;
  const int* bw = (const int*)(sWo8 + (size_t)n*192);
  __builtin_prefetch(bw, 0, 0);
  #pragma unroll 1
  for (int kc = 0; kc < 3; ++kc){
    AI32 a, bm;
    #pragma unroll
    for (int v = 0; v < 8; ++v){
      int word = 0;
      #pragma unroll
      for (int p = 0; p < 4; ++p){
        int kk = kc*64 + a8_kbase(v, hh) + p;
        int s = sgni(y2[(size_t)m*192 + kk]);
        word |= (s & 0xff) << (8*p);
      }
      a.i[v] = word;
    }
    #pragma unroll
    for (int v = 0; v < 8; ++v)
      bm.i[v] = bw[(kc*64 + b8_kbase(v, hh)) >> 2];
    WMMA_IU8(acc.v, a.v, bm.v);
  }
  float scv = s_out[n], bb = b_out[n];
  #pragma unroll
  for (int j = 0; j < 8; ++j){
    int mm = tm*16 + j + 8*hh;
    out[(size_t)mm*96 + n] = (float)acc.i[j]*scv + bb;
  }
}

extern "C" void kernel_launch(void* const* d_in, const int* in_sizes, int n_in,
                              void* d_out, int out_size, void* d_ws, size_t ws_size,
                              hipStream_t stream){
  (void)in_sizes; (void)n_in; (void)out_size; (void)ws_size;
  const float* x      = (const float*)d_in[0];
  const float* W_in   = (const float*)d_in[1];
  const float* b_in   = (const float*)d_in[2];
  const float* s_in   = (const float*)d_in[3];
  const float* move0  = (const float*)d_in[4];
  const float* conv_W = (const float*)d_in[5];
  const float* conv_b = (const float*)d_in[6];
  const float* rp_b0  = (const float*)d_in[7];
  const float* prelu  = (const float*)d_in[8];
  const float* rp_b1  = (const float*)d_in[9];
  const float* Wl     = (const float*)d_in[10];
  const float* bl     = (const float*)d_in[11];
  const float* sl     = (const float*)d_in[12];
  const float* Wd     = (const float*)d_in[13];
  const float* bd     = (const float*)d_in[14];
  const float* sd     = (const float*)d_in[15];
  const float* dt_b   = (const float*)d_in[16];
  const float* A_logs = (const float*)d_in[17];
  const float* Ds     = (const float*)d_in[18];
  const float* ln_w   = (const float*)d_in[19];
  const float* ln_b   = (const float*)d_in[20];
  const float* W_out  = (const float*)d_in[21];
  const float* b_out  = (const float*)d_in[22];
  const float* s_out  = (const float*)d_in[23];

  char* w = (char*)d_ws; size_t off = 0;
  auto carve = [&](size_t bytes)->void*{
    void* p = w + off; off = (off + bytes + 255) & ~(size_t)255; return p;
  };
  h16*         sWin  = (h16*)        carve(36864u*2);
  signed char* sWl8  = (signed char*)carve(29184u);
  signed char* sWo8  = (signed char*)carve(18432u);
  signed char* sCW8  = (signed char*)carve(331776u);
  float*       sc    = (float*)      carve(192u*4);
  float*       wdeff = (float*)      carve(4608u*4);
  float*       xp    = (float*)      carve(3145728ull*4);
  float*       z     = (float*)      carve(3145728ull*4);
  float*       xconv = (float*)      carve(3145728ull*4);
  float*       xdbl  = (float*)      carve(2490368ull*4);
  float*       dts   = (float*)      carve(12582912ull*4);
  float*       ycomb = (float*)      carve(3145728ull*4);
  float*       y2    = (float*)      carve(3145728ull*4);

  kprep_rowsign_f16<<<6, 64, 0, stream>>>(W_in,  sWin, 384, 96);
  kprep_rowsign_i8 <<<3, 64, 0, stream>>>(Wl,    sWl8, 152, 192);
  kprep_rowsign_i8 <<<2, 64, 0, stream>>>(W_out, sWo8,  96, 192);
  kprep_conv       <<<3, 64, 0, stream>>>(conv_W, sCW8, sc);
  kprep_wd         <<<12, 64, 0, stream>>>(Wd, sd, wdeff);

  k1_in_gemm<<<24576, 32, 0, stream>>>(x, sWin, s_in, b_in, xp, z);
  k2_conv   <<<1024, 384, 0, stream>>>(xp, sCW8, sc, move0, conv_b, rp_b0, prelu, rp_b1, xconv);
  k3_xdbl   <<<12288, 32, 0, stream>>>(xconv, sWl8, sl, bl, xdbl);
  k4_dts    <<<49152, 256, 0, stream>>>(xdbl, wdeff, bd, dt_b, dts);
  k5_zero   <<<12288, 256, 0, stream>>>(ycomb);
  k6_scan   <<<192, 256, 0, stream>>>(dts, xdbl, xconv, A_logs, Ds, ycomb);
  k7_ln     <<<16384, 192, 0, stream>>>(ycomb, z, ln_w, ln_b, y2);
  k8_out    <<<6144, 32, 0, stream>>>(y2, sWo8, s_out, b_out, (float*)d_out);
}
